// MODEL2_14920716387044
// MI455X (gfx1250) — matrix operand order repaired and checked by reading, not compiled, not run on hardware
//
#include <hip/hip_runtime.h>

#define BB   8
#define SS   4096
#define HH   8
#define DKK  64
#define DMM  512
#define DFFF 1024
#define ROWS (BB * SS)          // 32768 token rows

typedef __attribute__((ext_vector_type(16))) __bf16 v16bf;
typedef __attribute__((ext_vector_type(8)))  __bf16 v8bf;
typedef __attribute__((ext_vector_type(8)))  float  v8f;
typedef __attribute__((ext_vector_type(4)))  int    v4i;
typedef __attribute__((address_space(1)))    v4i    as1_v4i;  // global
typedef __attribute__((address_space(3)))    v4i    as3_v4i;  // LDS

#define DEVINL __device__ __forceinline__

// ---------------- WMMA fragment helpers (16x16x32 bf16, ISA layout) ---------
// A 16x32 (MxK): lanes 0-15 row M=lane, K={0..7,16..23}; lanes 16-31 row M=lane-16,
// K={8..15,24..31}.  Two contiguous 16-byte chunks per lane.
DEVINL v16bf combine_a(v8bf lo, v8bf hi) {
  v16bf r;
#pragma unroll
  for (int i = 0; i < 8; ++i) { r[i] = lo[i]; r[i + 8] = hi[i]; }
  return r;
}

DEVINL v16bf load_a_frag(const __bf16* __restrict__ A, int lda, int lane) {
  int mr = lane & 15;
  int kh = (lane >> 4) << 3;               // 0 or 8
  const __bf16* p = A + (size_t)mr * lda + kh;
  return combine_a(*(const v8bf*)p, *(const v8bf*)(p + 16));
}

// B 32x16 (KxN): lane = N column; lanes 0-15 hold K={0..7,16..23}, lanes 16-31
// hold K={8..15,24..31} -> two 16-byte chunks per lane from K-major W[n, k].
DEVINL v16bf load_b_frag(const __bf16* __restrict__ W, int ldb, int lane) {
  int nc = lane & 15;
  int kh = (lane >> 4) << 3;               // 0 or 8
  const __bf16* p = W + (size_t)nc * ldb + kh;
  return combine_a(*(const v8bf*)p, *(const v8bf*)(p + 16));
}

// ---- gfx1250 async global->LDS staging (ASYNCcnt path), with safe fallback --
DEVINL void stage16_to_lds(__bf16* dstLDS, const __bf16* src) {
#if __has_builtin(__builtin_amdgcn_global_load_async_to_lds_b128)
  __builtin_amdgcn_global_load_async_to_lds_b128(
      (as1_v4i*)src, (as3_v4i*)dstLDS, 0, 0);
#else
  *(v8bf*)dstLDS = *(const v8bf*)src;      // global_load_b128 + ds_store_b128
#endif
}

DEVINL void async_stage_wait() {
#if __has_builtin(__builtin_amdgcn_s_wait_asynccnt)
  __builtin_amdgcn_s_wait_asynccnt(0);
#endif
}

// ---------------- bf16 WMMA GEMM:  C[M,N] = act(A[M,K] @ W[N,K]^T + b) -------
// Block = 256 threads (8 waves) -> 256x64 tile; each wave owns a 32x64 tile
// (2 A fragments, 8 WMMAs per k-step).  The 64x32 B k-slice is async-staged
// once per k-step into ping-pong LDS and shared by all 8 waves; A fragments
// are double-buffered through registers so WMMAs never wait on current loads.
// act: 0=none, 1=relu, 2=elu(x)+1 (linear-attention feature map).
__global__ __launch_bounds__(256)
void gemm_bf16_wmma(const __bf16* __restrict__ A,
                    const __bf16* __restrict__ W,
                    const float*  __restrict__ bias,
                    float*        __restrict__ Cf,
                    __bf16*       __restrict__ Cb,
                    int M, int N, int K, int act) {
  __shared__ __bf16 sB[2][64 * 32];        // 2 x 4KB ping-pong
  int t    = threadIdx.x;
  int lane = t & 31;
  int wave = t >> 5;                       // 0..7
  int n0 = blockIdx.x * 64;
  int m0 = blockIdx.y * 256 + wave * 32;

  // B staging: thread t moves 16B of W row (n0 + t/4), cols (t%4)*8 .. +7
  int bn = t >> 2;
  int bk = (t & 3) * 8;
  const __bf16* gB = W + (size_t)(n0 + bn) * K + bk;
  __bf16* lB0 = &sB[0][bn * 32 + bk];
  __bf16* lB1 = &sB[1][bn * 32 + bk];

  // A fragment addressing for this wave's two 16-row tiles
  int mr  = lane & 15;
  int kh8 = (lane >> 4) << 3;
  const __bf16* gA0 = A + (size_t)(m0 + mr) * K + kh8;
  const __bf16* gA1 = A + (size_t)(m0 + 16 + mr) * K + kh8;

  v8f z8 = {0.f, 0.f, 0.f, 0.f, 0.f, 0.f, 0.f, 0.f};
  v8f acc[2][4] = {{z8, z8, z8, z8}, {z8, z8, z8, z8}};

  stage16_to_lds(lB0, gB);                 // prime buffer 0 with k=0 slice
  // prime A registers for k=0
  v8bf a0lo = *(const v8bf*)(gA0);
  v8bf a0hi = *(const v8bf*)(gA0 + 16);
  v8bf a1lo = *(const v8bf*)(gA1);
  v8bf a1hi = *(const v8bf*)(gA1 + 16);

  int p = 0;
  for (int k0 = 0; k0 < K; k0 += 32) {
    async_stage_wait();                    // staged writes for buffer p landed
    __syncthreads();                       // visible to all waves; p^1 reads done
    bool more = (k0 + 32 < K);
    if (more)                              // stage next B slice while computing
      stage16_to_lds(p ? lB0 : lB1, gB + k0 + 32);

    v16bf a0 = combine_a(a0lo, a0hi);      // fragments for THIS k-step
    v16bf a1 = combine_a(a1lo, a1hi);
    if (more) {                            // register double-buffer: next A
      a0lo = *(const v8bf*)(gA0 + k0 + 32);
      a0hi = *(const v8bf*)(gA0 + k0 + 48);
      a1lo = *(const v8bf*)(gA1 + k0 + 32);
      a1hi = *(const v8bf*)(gA1 + k0 + 48);
    }
    if (k0 + 64 < K)                       // prefetch A beyond that (b8 hint)
      __builtin_prefetch((const void*)(gA0 + k0 + 64), 0, 1);

    const __bf16* sb = sB[p];
    v16bf b0 = load_b_frag(sb +  0 * 32, 32, lane);   // issue all ds_loads
    v16bf b1 = load_b_frag(sb + 16 * 32, 32, lane);   // before the WMMA cluster
    v16bf b2 = load_b_frag(sb + 32 * 32, 32, lane);
    v16bf b3 = load_b_frag(sb + 48 * 32, 32, lane);

    acc[0][0] = __builtin_amdgcn_wmma_f32_16x16x32_bf16(false, a0, false, b0, (short)0, acc[0][0], false, false);
    acc[0][1] = __builtin_amdgcn_wmma_f32_16x16x32_bf16(false, a0, false, b1, (short)0, acc[0][1], false, false);
    acc[0][2] = __builtin_amdgcn_wmma_f32_16x16x32_bf16(false, a0, false, b2, (short)0, acc[0][2], false, false);
    acc[0][3] = __builtin_amdgcn_wmma_f32_16x16x32_bf16(false, a0, false, b3, (short)0, acc[0][3], false, false);
    acc[1][0] = __builtin_amdgcn_wmma_f32_16x16x32_bf16(false, a1, false, b0, (short)0, acc[1][0], false, false);
    acc[1][1] = __builtin_amdgcn_wmma_f32_16x16x32_bf16(false, a1, false, b1, (short)0, acc[1][1], false, false);
    acc[1][2] = __builtin_amdgcn_wmma_f32_16x16x32_bf16(false, a1, false, b2, (short)0, acc[1][2], false, false);
    acc[1][3] = __builtin_amdgcn_wmma_f32_16x16x32_bf16(false, a1, false, b3, (short)0, acc[1][3], false, false);
    p ^= 1;
  }

  // C/D layout: lane 0-15 -> N=lane, M=vgpr r; lane 16-31 -> N=lane-16, M=8+r
  int rbase = (lane >> 4) << 3;
  int col   = lane & 15;
#pragma unroll
  for (int i = 0; i < 2; ++i) {
#pragma unroll
    for (int j = 0; j < 4; ++j) {
      int n = n0 + j * 16 + col;
      float bia = bias ? bias[n] : 0.f;
#pragma unroll
      for (int r = 0; r < 8; ++r) {
        int m = m0 + i * 16 + rbase + r;
        float v = acc[i][j][r] + bia;
        if (act == 1)      v = fmaxf(v, 0.f);
        else if (act == 2) v = (v > 0.f) ? (v + 1.f) : __expf(v);   // elu(x)+1
        size_t off = (size_t)m * N + n;
        if (Cf) Cf[off] = v;
        if (Cb) Cb[off] = (__bf16)v;
      }
    }
  }
}

// ---------------- Attention output: out[s,m] = Z[s] * sum_d phiQ[s,d]*KV[d,m]
// grid = (S/16, H, B); K=64 so the whole KV tile stays in registers.
__global__ void attn_out_wmma(const __bf16* __restrict__ phiQ,   // [B*S, 512]
                              const __bf16* __restrict__ KVt,    // [B*H, 64*64]
                              const float*  __restrict__ Z,      // [B*S, H]
                              __bf16*       __restrict__ attn) { // [B*S, 512]
  int lane = threadIdx.x;
  int s0 = blockIdx.x * 16;
  int h  = blockIdx.y;
  int b  = blockIdx.z;
  const __bf16* Abase = phiQ + (size_t)b * SS * DMM + (size_t)h * DKK; // +s*512+d
  const __bf16* Bbase = KVt + (size_t)(b * HH + h) * DKK * DKK;        // [m][d]
  v8f z8 = {0.f, 0.f, 0.f, 0.f, 0.f, 0.f, 0.f, 0.f};
  v8f acc[4] = {z8, z8, z8, z8};
#pragma unroll
  for (int k0 = 0; k0 < DKK; k0 += 32) {
    v16bf a = load_a_frag(Abase + (size_t)s0 * DMM + k0, DMM, lane);
    v16bf b0 = load_b_frag(Bbase + (size_t) 0 * DKK + k0, DKK, lane);
    v16bf b1 = load_b_frag(Bbase + (size_t)16 * DKK + k0, DKK, lane);
    v16bf b2 = load_b_frag(Bbase + (size_t)32 * DKK + k0, DKK, lane);
    v16bf b3 = load_b_frag(Bbase + (size_t)48 * DKK + k0, DKK, lane);
    acc[0] = __builtin_amdgcn_wmma_f32_16x16x32_bf16(false, a, false, b0, (short)0, acc[0], false, false);
    acc[1] = __builtin_amdgcn_wmma_f32_16x16x32_bf16(false, a, false, b1, (short)0, acc[1], false, false);
    acc[2] = __builtin_amdgcn_wmma_f32_16x16x32_bf16(false, a, false, b2, (short)0, acc[2], false, false);
    acc[3] = __builtin_amdgcn_wmma_f32_16x16x32_bf16(false, a, false, b3, (short)0, acc[3], false, false);
  }
  int rbase = (lane >> 4) << 3;
  int col   = lane & 15;
#pragma unroll
  for (int j = 0; j < 4; ++j) {
#pragma unroll
    for (int r = 0; r < 8; ++r) {
      int s = s0 + rbase + r;
      float zz = Z[((size_t)b * SS + s) * HH + h];
      attn[((size_t)b * SS + s) * DMM + h * DKK + j * 16 + col] =
          (__bf16)(acc[j][r] * zz);
    }
  }
}

// ---------------- KV reduction: KVt[bh][m][d] = sum_s phiK[s,d] * V[s,m] ------
__global__ void kv_reduce(const __bf16* __restrict__ phiK,
                          const __bf16* __restrict__ V,
                          float* __restrict__ KVt) {
  int bh = blockIdx.x;                 // B*H = 64 blocks
  int t  = threadIdx.x;                // 256 threads
  int b = bh / HH, h = bh % HH;
  const __bf16* kb = phiK + (size_t)b * SS * DMM + (size_t)h * DKK;
  const __bf16* vb = V    + (size_t)b * SS * DMM + (size_t)h * DKK;
  __shared__ float sK[4 * 64];
  __shared__ float sV[4 * 64];
  float acc[16];
#pragma unroll
  for (int i = 0; i < 16; ++i) acc[i] = 0.f;
  int si = t >> 6, dd = t & 63;
  for (int s0 = 0; s0 < SS; s0 += 4) {
    __syncthreads();
    sK[t] = (float)kb[(size_t)(s0 + si) * DMM + dd];
    sV[t] = (float)vb[(size_t)(s0 + si) * DMM + dd];
    __syncthreads();
#pragma unroll
    for (int ss = 0; ss < 4; ++ss) {
#pragma unroll
      for (int i = 0; i < 16; ++i) {
        int p = t * 16 + i;
        acc[i] += sK[ss * 64 + (p & 63)] * sV[ss * 64 + (p >> 6)];
      }
    }
  }
#pragma unroll
  for (int i = 0; i < 16; ++i) {
    int p = t * 16 + i;
    KVt[((size_t)bh * DKK + (p >> 6)) * DKK + (p & 63)] = acc[i];
  }
}

// ---------------- Ksum[bh][d] = sum_s phiK[s,d] ------------------------------
__global__ void ksum_kernel(const __bf16* __restrict__ phiK, float* __restrict__ Ksum) {
  int bh = blockIdx.x;
  int d  = threadIdx.x;               // 64 threads
  int b = bh / HH, h = bh % HH;
  const __bf16* base = phiK + (size_t)b * SS * DMM + (size_t)h * DKK + d;
  float acc = 0.f;
  for (int s = 0; s < SS; ++s) acc += (float)base[(size_t)s * DMM];
  Ksum[(size_t)bh * DKK + d] = acc;
}

// ---------------- Z[b,s,h] = 1 / (phiQ . Ksum + eps) -------------------------
__global__ void z_kernel(const __bf16* __restrict__ phiQ,
                         const float* __restrict__ Ksum,
                         float* __restrict__ Z) {
  int i = blockIdx.x * blockDim.x + threadIdx.x;   // B*S*H
  if (i >= ROWS * HH) return;
  int h = i % HH;
  int row = i / HH;
  int b = row / SS;
  const __bf16* q = phiQ + (size_t)row * DMM + h * DKK;
  const float* ks = Ksum + ((size_t)b * HH + h) * DKK;
  float acc = 0.f;
#pragma unroll 8
  for (int d = 0; d < DKK; ++d) acc += (float)q[d] * ks[d];
  Z[i] = 1.0f / (acc + 1e-6f);
}

// ---------------- residual add + LayerNorm (block per row of 512) ------------
__global__ void add_ln_kernel(const float* __restrict__ xin,
                              const float* __restrict__ delta,
                              const float* __restrict__ g,
                              const float* __restrict__ bta,
                              float* __restrict__ xout,
                              __bf16* __restrict__ xbout) {
  int row = blockIdx.x;
  int t   = threadIdx.x;               // 256 threads, 2 elems each
  size_t base = (size_t)row * DMM;
  float v0 = xin[base + t]       + (delta ? delta[base + t]       : 0.f);
  float v1 = xin[base + t + 256] + (delta ? delta[base + t + 256] : 0.f);
  __shared__ float red[256];
  red[t] = v0 + v1;
  __syncthreads();
  for (int o = 128; o > 0; o >>= 1) { if (t < o) red[t] += red[t + o]; __syncthreads(); }
  float mu = red[0] * (1.0f / DMM);
  __syncthreads();
  float d0 = v0 - mu, d1 = v1 - mu;
  red[t] = d0 * d0 + d1 * d1;
  __syncthreads();
  for (int o = 128; o > 0; o >>= 1) { if (t < o) red[t] += red[t + o]; __syncthreads(); }
  float rs = rsqrtf(red[0] * (1.0f / DMM) + 1e-5f);
  float o0 = d0 * rs * g[t]       + bta[t];
  float o1 = d1 * rs * g[t + 256] + bta[t + 256];
  xout[base + t]       = o0;
  xout[base + t + 256] = o1;
  if (xbout) { xbout[base + t] = (__bf16)o0; xbout[base + t + 256] = (__bf16)o1; }
}

// ---------------- head: out[row] = x[row,:] . hw + hb ------------------------
__global__ void head_kernel(const float* __restrict__ x,
                            const float* __restrict__ hw,
                            const float* __restrict__ hb,
                            float* __restrict__ out) {
  int row = blockIdx.x;
  int t   = threadIdx.x;
  size_t base = (size_t)row * DMM;
  __shared__ float red[256];
  red[t] = x[base + t] * hw[t] + x[base + t + 256] * hw[t + 256];
  __syncthreads();
  for (int o = 128; o > 0; o >>= 1) { if (t < o) red[t] += red[t + o]; __syncthreads(); }
  if (t == 0) out[row] = red[0] + hb[0];
}

// ---------------- elementwise converters ------------------------------------
__global__ void init_x_kernel(const float* __restrict__ src,
                              float* __restrict__ xf, __bf16* __restrict__ xb, int n) {
  int i = blockIdx.x * blockDim.x + threadIdx.x;
  if (i >= n) return;
  float v = src[i];
  xf[i] = v;
  xb[i] = (__bf16)v;
}

__global__ void cvt_f32_bf16(const float* __restrict__ src, __bf16* __restrict__ dst, int n) {
  int i = blockIdx.x * blockDim.x + threadIdx.x;
  if (i < n) dst[i] = (__bf16)src[i];
}

// ---------------- host orchestration ----------------------------------------
extern "C" void kernel_launch(void* const* d_in, const int* in_sizes, int n_in,
                              void* d_out, int out_size, void* d_ws, size_t ws_size,
                              hipStream_t stream) {
  (void)in_sizes; (void)n_in; (void)out_size; (void)ws_size;
  const float* emb = (const float*)d_in[0];
  // per-layer params: wq bq wk bk wv bv wo bo ln1g ln1b w1 b1 w2 b2 ln2g ln2b lnfg lnfb (18)
  auto P = [&](int layer, int idx) -> const float* {
    return (const float*)d_in[1 + layer * 18 + idx];
  };
  const float* head_w = (const float*)d_in[1 + 2 * 18];
  const float* head_b = (const float*)d_in[2 + 2 * 18];

  char* wsb = (char*)d_ws;
  auto alloc = [&](size_t bytes) -> char* {
    char* p = wsb;
    wsb += (bytes + 255) & ~(size_t)255;
    return p;
  };
  float*  X     = (float*) alloc((size_t)ROWS * DMM * 4);      // 64 MB
  __bf16* XB    = (__bf16*)alloc((size_t)ROWS * DMM * 2);      // 32 MB
  float*  T0    = (float*) alloc((size_t)ROWS * DMM * 4);      // 64 MB gemm f32 out
  __bf16* H1    = (__bf16*)alloc((size_t)ROWS * DFFF * 2);     // 64 MB
  __bf16* PHIQ  = (__bf16*)alloc((size_t)ROWS * DMM * 2);      // 32 MB
  __bf16* PHIK  = (__bf16*)alloc((size_t)ROWS * DMM * 2);      // 32 MB
  __bf16* VV    = (__bf16*)alloc((size_t)ROWS * DMM * 2);      // 32 MB
  __bf16* ATTN  = (__bf16*)alloc((size_t)ROWS * DMM * 2);      // 32 MB
  float*  KVT_F = (float*) alloc((size_t)BB * HH * DKK * DKK * 4);
  __bf16* KVT_B = (__bf16*)alloc((size_t)BB * HH * DKK * DKK * 2);
  float*  KSUM  = (float*) alloc((size_t)BB * HH * DKK * 4);
  float*  Zbuf  = (float*) alloc((size_t)ROWS * HH * 4);
  __bf16* WB    = (__bf16*)alloc((size_t)DFFF * DMM * 2);      // weight scratch

  const int NE = ROWS * DMM;                                   // 16.7M
  init_x_kernel<<<(NE + 255) / 256, 256, 0, stream>>>(emb, X, XB, NE);

  auto cvt = [&](const float* s, __bf16* d, int n) {
    cvt_f32_bf16<<<(n + 255) / 256, 256, 0, stream>>>(s, d, n);
  };
  auto gemm = [&](const __bf16* A, const __bf16* W, const float* bias,
                  float* Cf, __bf16* Cb, int M, int N, int K, int act) {
    gemm_bf16_wmma<<<dim3(N / 64, M / 256), 256, 0, stream>>>(A, W, bias, Cf, Cb, M, N, K, act);
  };

  for (int l = 0; l < 2; ++l) {
    // q, k with feature map elu(x)+1; v plain
    cvt(P(l, 0), WB, DMM * DMM);
    gemm(XB, WB, P(l, 1), nullptr, PHIQ, ROWS, DMM, DMM, 2);
    cvt(P(l, 2), WB, DMM * DMM);
    gemm(XB, WB, P(l, 3), nullptr, PHIK, ROWS, DMM, DMM, 2);
    cvt(P(l, 4), WB, DMM * DMM);
    gemm(XB, WB, P(l, 5), nullptr, VV, ROWS, DMM, DMM, 0);

    // linear attention state
    kv_reduce<<<BB * HH, 256, 0, stream>>>(PHIK, VV, KVT_F);
    ksum_kernel<<<BB * HH, 64, 0, stream>>>(PHIK, KSUM);
    cvt(KVT_F, KVT_B, BB * HH * DKK * DKK);
    z_kernel<<<(ROWS * HH + 255) / 256, 256, 0, stream>>>(PHIQ, KSUM, Zbuf);
    attn_out_wmma<<<dim3(SS / 16, HH, BB), 32, 0, stream>>>(PHIQ, KVT_B, Zbuf, ATTN);

    // output projection + post-norm
    cvt(P(l, 6), WB, DMM * DMM);
    gemm(ATTN, WB, P(l, 7), T0, nullptr, ROWS, DMM, DMM, 0);
    add_ln_kernel<<<ROWS, 256, 0, stream>>>(X, T0, P(l, 8), P(l, 9), X, XB);

    // FFN + post-norm
    cvt(P(l, 10), WB, DFFF * DMM);
    gemm(XB, WB, P(l, 11), nullptr, H1, ROWS, DFFF, DMM, 1);
    cvt(P(l, 12), WB, DMM * DFFF);
    gemm(H1, WB, P(l, 13), T0, nullptr, ROWS, DMM, DFFF, 0);
    add_ln_kernel<<<ROWS, 256, 0, stream>>>(X, T0, P(l, 14), P(l, 15), X, XB);
  }

  // final LayerNorm (params[-1].lnf) + head
  add_ln_kernel<<<ROWS, 256, 0, stream>>>(X, nullptr, P(1, 16), P(1, 17), X, nullptr);
  head_kernel<<<ROWS, 256, 0, stream>>>(X, head_w, head_b, (float*)d_out);
}
